// backbone_41644002902564
// MI455X (gfx1250) — matrix operand order repaired and checked by reading, not compiled, not run on hardware
//
#include <hip/hip_runtime.h>
#include <hip/hip_bf16.h>

// ---------------- CDNA5 WMMA types & helpers ----------------
typedef __attribute__((ext_vector_type(16))) __bf16 v16bf;
typedef __attribute__((ext_vector_type(8)))  float  v8f;
typedef __attribute__((ext_vector_type(4)))  unsigned int v4u;
typedef __attribute__((ext_vector_type(8)))  int v8i;
typedef __attribute__((ext_vector_type(4)))  int v4i;

static __device__ __forceinline__ unsigned short bf16bits(float f) {
  unsigned u = __float_as_uint(f);
  u += 0x7FFFu + ((u >> 16) & 1u);          // round-to-nearest-even
  return (unsigned short)(u >> 16);
}

union FragU { v16bf v; v4u q[2]; unsigned short u[16]; };

// A fragment: matrix stored M x K row-major (bf16 bits in ushort), tile (r0,k0).
// Lane l: row = r0 + (l&15); v[0..7] = K kb..kb+7, v[8..15] = K kb+16..kb+23,
// kb = k0 + (l>>4)*8.   (CDNA5 ISA 7.12.2, 16-bit A 16x32)
static __device__ __forceinline__ v16bf load_frag_a(const unsigned short* __restrict__ src,
                                                    int r0, int k0, int ld, int lane) {
  FragU f;
  const unsigned short* p = src + (r0 + (lane & 15)) * ld + k0 + ((lane >> 4) << 3);
  f.q[0] = *(const v4u*)(p);
  f.q[1] = *(const v4u*)(p + 16);
  return f.v;
}

// B fragment: B is K x N, stored here as N x K row-major ("TN" GEMM).
// Lane l: col n = n0 + (l&15); v[0..7] = K kb..kb+7, v[8..15] = K kb+16..kb+23, kb = k0 + (l>>4)*8.
static __device__ __forceinline__ v16bf load_frag_b(const unsigned short* __restrict__ src,
                                                    int n0, int k0, int ld, int lane) {
  FragU f;
  const unsigned short* p = src + (n0 + (lane & 15)) * ld + k0 + ((lane >> 4) << 3);
  f.q[0] = *(const v4u*)(p);
  f.q[1] = *(const v4u*)(p + 16);
  return f.v;
}

#define WMMA_BF16(acc, a, b) \
  (acc) = __builtin_amdgcn_wmma_f32_16x16x32_bf16(false, (a), false, (b), (short)0, (acc), false, false)

// ---------------- Tensor Data Mover: 1-D bulk copy global -> LDS ----------------
#if __has_builtin(__builtin_amdgcn_tensor_load_to_lds)
#define HAVE_TDM 1
#else
#define HAVE_TDM 0
#endif

#if HAVE_TDM
// n8 = number of 8-byte elements (data_size=3). Builds D# per CDNA5 ISA ch.8:
// group0: count=1 | lds_addr | global_addr[56:0] | type=2
// group1: data_size=3; tensor_dim0=tile_dim0=n8; tensor_dim1=1; dim0_stride=n8
static __device__ __forceinline__ void tdm_copy_to_lds(const void* lds_dst, const void* gsrc,
                                                       unsigned n8) {
  unsigned long long ga = (unsigned long long)(size_t)gsrc;
  unsigned lds_addr = (unsigned)(size_t)lds_dst;  // LDS aperture: low 32 bits = byte offset
  v4u g0;
  g0[0] = 1u;                                              // count=1, user descriptor
  g0[1] = lds_addr;
  g0[2] = (unsigned)(ga & 0xFFFFFFFFu);
  g0[3] = (unsigned)((ga >> 32) & 0x01FFFFFFu) | (2u << 30);  // type=2 ("image")
  v8i g1;
  g1[0] = (int)(3u << 16);                                 // workgroup_mask=0, data_size=3 (8B)
  g1[1] = (int)((n8 & 0xFFFFu) << 16);                     // tensor_dim0[15:0] in bits[63:48]
  g1[2] = (int)(((n8 >> 16) & 0xFFFFu) | (1u << 16));      // tensor_dim0[31:16], tensor_dim1=1
  g1[3] = (int)((n8 & 0xFFFFu) << 16);                     // tile_dim0 = n8
  g1[4] = 1;                                               // tile_dim1=1, tile_dim2=0
  g1[5] = (int)n8;                                         // tensor_dim0_stride[31:0]
  g1[6] = 0;                                               // stride hi / dim1_stride lo
  g1[7] = 0;
  v4i z = {0, 0, 0, 0};
#if __clang_major__ >= 23
  v8i z8 = {0, 0, 0, 0, 0, 0, 0, 0};
  __builtin_amdgcn_tensor_load_to_lds(g0, g1, z, z, z8, 0);
#else
  __builtin_amdgcn_tensor_load_to_lds(g0, g1, z, z, 0);
#endif
}
#endif

// dims
#define NS 64   // samples
#define CC 512  // in channels
#define TT 64   // time
#define PP 2016 // pairs

// ---------------- kernel 0: weight/x prep (f32 -> bf16, transposes) ----------------
__global__ void k0_prep(const float* __restrict__ x, const float* __restrict__ wconv,
                        const float* __restrict__ wc1,
                        unsigned short* __restrict__ wA,   // [256][512]
                        unsigned short* __restrict__ wC,   // [64][256] rows 0-31 even, 32-63 odd
                        unsigned short* __restrict__ xT) { // [n][t][c] = x[n][c][t]
  int g = blockIdx.x * blockDim.x + threadIdx.x;
  int stride = gridDim.x * blockDim.x;
  for (int idx = g; idx < 256 * 512; idx += stride) wA[idx] = bf16bits(wconv[idx]);
  for (int idx = g; idx < 64 * 256; idx += stride) {
    int o = idx >> 8, c = idx & 255;
    int src = (o < 32) ? (o * 512 + 2 * c) : ((o - 32) * 512 + 2 * c + 1);
    wC[idx] = bf16bits(wc1[src]);
  }
  for (int idx = g; idx < NS * CC * TT; idx += stride) {
    int n = idx >> 15, c = (idx >> 6) & 511, t = idx & 63;
    xT[n * (TT * CC) + t * CC + c] = bf16bits(x[idx]);
  }
}

// ---------------- kernel 1: per-sample f = W@x + b, G = Wc@f ----------------
// one block per sample, 256 threads = 8 waves.
// Dynamic LDS: xT[64][512] bf16 @0 (64KB), fT[64][256] bf16 @65536 (32KB)
__global__ void k1_feat(const unsigned short* __restrict__ xT, const float* __restrict__ b_conv,
                        const unsigned short* __restrict__ wA, const unsigned short* __restrict__ wC,
                        unsigned short* __restrict__ F,   // [n][256][64] (c-major)
                        unsigned short* __restrict__ G) { // [n][64][64]
  extern __shared__ unsigned char smem[];
  unsigned short* sX = (unsigned short*)smem;            // [t][c] 64 x 512
  unsigned short* fT = (unsigned short*)(smem + 65536);  // [t][c'] 64 x 256
  const int n = blockIdx.x;
  const int tid = threadIdx.x, lane = tid & 31, wave = tid >> 5;
  const unsigned short* xTn = xT + n * (TT * CC);

  // stage x[n]^T into LDS (64KB) -- TDM if available
#if HAVE_TDM
  if (tid == 0) {
    tdm_copy_to_lds(sX, xTn, (TT * CC * 2) / 8);   // 8192 x 8B
    __builtin_amdgcn_s_wait_tensorcnt(0);
  }
#else
  {
    const v4u* src = (const v4u*)xTn;
    v4u* dst = (v4u*)sX;
    for (int q = tid; q < (TT * CC) / 8; q += 256) dst[q] = src[q];
  }
#endif
  __syncthreads();

  // f-GEMM: f(256x64) = W(256x512) @ x[n];  A = W (row-major), B = sX rows (t-major)
#pragma unroll 1
  for (int s = 0; s < 8; ++s) {                 // contiguous tile range per wave
    int tt = wave * 8 + s;                      // 16 row-tiles x 4 col-tiles
    int ro = tt >> 2, co = tt & 3;
    v8f acc = {0.f, 0.f, 0.f, 0.f, 0.f, 0.f, 0.f, 0.f};
#pragma unroll 4
    for (int k0 = 0; k0 < 512; k0 += 32) {
      v16bf a = load_frag_a(wA, ro * 16, k0, 512, lane);
      v16bf b = load_frag_b(sX, co * 16, k0, 512, lane);
      WMMA_BF16(acc, a, b);
    }
    int t = co * 16 + (lane & 15);
    int mbase = (lane >> 4) << 3;
    union { unsigned short u[8]; v4u q[2]; } tmp;
#pragma unroll
    for (int r = 0; r < 8; ++r) {
      int o = ro * 16 + mbase + r;
      unsigned short h = bf16bits(acc[r] + b_conv[o]);
      F[n * (256 * 64) + o * 64 + t] = h;   // c-major for pair-kernel B operand
      tmp.u[r] = h;
    }
    *(v4u*)(fT + t * 256 + ro * 16 + mbase) = tmp.q[0];  // t-major, contiguous 8
  }
  __syncthreads();

  // G-GEMM: G(64x64) = Wc(64x256) @ f ; B operand = fT rows (t-major)
#pragma unroll 1
  for (int s = 0; s < 2; ++s) {
    int tt = wave * 2 + s;                      // 4x4 tiles
    int ro = tt >> 2, co = tt & 3;
    v8f acc = {0.f, 0.f, 0.f, 0.f, 0.f, 0.f, 0.f, 0.f};
#pragma unroll 4
    for (int k0 = 0; k0 < 256; k0 += 32) {
      v16bf a = load_frag_a(wC, ro * 16, k0, 256, lane);
      v16bf b = load_frag_b(fT, co * 16, k0, 256, lane);
      WMMA_BF16(acc, a, b);
    }
    int t = co * 16 + (lane & 15);
    int mbase = (lane >> 4) << 3;
#pragma unroll
    for (int r = 0; r < 8; ++r) {
      int o = ro * 16 + mbase + r;
      G[n * 4096 + o * 64 + t] = bf16bits(acc[r]);
    }
  }
}

// ---------------- kernel 2: per-sample score = sigmoid(x . w_mlp + b) ----------------
__global__ void k2_score(const float* __restrict__ x, const float* __restrict__ w_mlp,
                         const float* __restrict__ b_mlp, float* __restrict__ out) {
  __shared__ float red[256];
  int n = blockIdx.x;
  const float* xn = x + n * (CC * TT);
  float s = 0.f;
  for (int idx = threadIdx.x; idx < CC * TT; idx += 256) s += xn[idx] * w_mlp[idx];
  red[threadIdx.x] = s;
  __syncthreads();
  for (int off = 128; off > 0; off >>= 1) {
    if (threadIdx.x < off) red[threadIdx.x] += red[threadIdx.x + off];
    __syncthreads();
  }
  if (threadIdx.x == 0) out[n] = 1.f / (1.f + __expf(-(red[0] + b_mlp[0])));
}

// ---------------- kernel 3: per-pair fused H-GEMM + pools + convs + FCs ----------------
// one block per pair, 256 threads = 8 waves.
// Dynamic LDS map (bytes): sH f32[32*128] @0 (16384); sG bf16[64*64] @16384 (8192);
// sF bf16[256*64] @24576 (32768). After H is built, sG/sF region is reused for the tail.
__global__ void k3_pair(const unsigned short* __restrict__ F, const unsigned short* __restrict__ G,
                        const float* __restrict__ b_c1,
                        const float* __restrict__ w_c2, const float* __restrict__ b_c2,
                        const float* __restrict__ w_fc1, const float* __restrict__ b_fc1,
                        const float* __restrict__ w_fc2, const float* __restrict__ b_fc2,
                        const float* __restrict__ w_fc3, const float* __restrict__ b_fc3,
                        float* __restrict__ out) {
  extern __shared__ unsigned char smem[];
  float* sH = (float*)smem;                              // 32 x 128
  unsigned short* sG = (unsigned short*)(smem + 16384);  // 64 x 64
  unsigned short* sF = (unsigned short*)(smem + 24576);  // 256 x 64
  float* sP1 = (float*)(smem + 16384);                   // reuse: 32 x 32
  float* sO2 = (float*)(smem + 16384 + 4096);            // 8 x 32
  float* sV  = (float*)(smem + 16384 + 4096 + 1024);     // 64
  float* sV1 = sV + 64;                                  // 32
  float* sV2 = sV1 + 32;                                 // 8

  const int p = blockIdx.x;
  int i = 0, rem = p;
  while (rem >= 63 - i) { rem -= 63 - i; ++i; }          // triu(k=1) order
  const int j = i + 1 + rem;
  const int tid = threadIdx.x, lane = tid & 31, wave = tid >> 5;

  // stage G_i (8KB) and F_j (32KB) into LDS via TDM
#if HAVE_TDM
  if (tid == 0) {
    tdm_copy_to_lds(sG, G + i * 4096, 4096 * 2 / 8);     // 1024 x 8B
    tdm_copy_to_lds(sF, F + j * 16384, 16384 * 2 / 8);   // 4096 x 8B
    __builtin_amdgcn_s_wait_tensorcnt(0);
  }
#else
  {
    const v4u* gsrc = (const v4u*)(G + i * 4096);
    v4u* gdst = (v4u*)sG;
    for (int q = tid; q < 512; q += 256) gdst[q] = gsrc[q];
    const v4u* fsrc = (const v4u*)(F + j * 16384);
    v4u* fdst = (v4u*)sF;
    for (int q = tid; q < 2048; q += 256) fdst[q] = fsrc[q];
  }
#endif
  __syncthreads();

  // H(32x128): H[o,d] = sum_t Ge[o,t]*Fj[d,t] + sum_t Go[o,t]*Fj[d+128,t]  (+ b_c1[o])
#pragma unroll 1
  for (int s = 0; s < 2; ++s) {
    int tt = wave * 2 + s;                               // 2 row-tiles x 8 col-tiles
    int ro = tt >> 3, co = tt & 7;
    v8f acc = {0.f, 0.f, 0.f, 0.f, 0.f, 0.f, 0.f, 0.f};
#pragma unroll
    for (int part = 0; part < 2; ++part) {
#pragma unroll
      for (int k0 = 0; k0 < 64; k0 += 32) {
        v16bf a = load_frag_a(sG, part * 32 + ro * 16, k0, 64, lane);
        v16bf b = load_frag_b(sF, part * 128 + co * 16, k0, 64, lane);
        WMMA_BF16(acc, a, b);
      }
    }
    int d = co * 16 + (lane & 15);
    int mbase = (lane >> 4) << 3;
#pragma unroll
    for (int r = 0; r < 8; ++r) {
      int o = ro * 16 + mbase + r;
      sH[o * 128 + d] = acc[r] + b_c1[o];
    }
  }
  __syncthreads();

  // maxpool2x2 on [32ch][32][4] -> sP1 [32ch][16][2]   (overwrites sG region)
  for (int q = tid; q < 1024; q += 256) {
    int c = q >> 5, ph = (q >> 1) & 15, pw = q & 1;
    const float* hr = sH + c * 128;
    int base = ph * 8 + pw * 2;
    sP1[q] = fmaxf(fmaxf(hr[base], hr[base + 1]), fmaxf(hr[base + 4], hr[base + 5]));
  }
  __syncthreads();

  // 1x1 conv 32->8 over 32 spatial positions + bias
  {
    int o2 = tid >> 5, s = tid & 31;
    float a = b_c2[o2];
#pragma unroll
    for (int c = 0; c < 32; ++c) a += w_c2[o2 * 32 + c] * sP1[c * 32 + s];
    sO2[o2 * 32 + s] = a;
  }
  __syncthreads();

  // maxpool2x2 on [8ch][16][2] -> v[64]
  if (tid < 64) {
    int o2 = tid >> 3, qh = tid & 7;
    const float* orow = sO2 + o2 * 32;
    sV[o2 * 8 + qh] = fmaxf(fmaxf(orow[4 * qh], orow[4 * qh + 1]),
                            fmaxf(orow[4 * qh + 2], orow[4 * qh + 3]));
  }
  __syncthreads();

  if (tid < 32) {  // fc1 64->32 + relu
    float a = b_fc1[tid];
#pragma unroll
    for (int k = 0; k < 64; ++k) a += w_fc1[tid * 64 + k] * sV[k];
    sV1[tid] = fmaxf(a, 0.f);
  }
  __syncthreads();
  if (tid < 8) {   // fc2 32->8 + relu
    float a = b_fc2[tid];
#pragma unroll
    for (int k = 0; k < 32; ++k) a += w_fc2[tid * 32 + k] * sV1[k];
    sV2[tid] = fmaxf(a, 0.f);
  }
  __syncthreads();
  if (tid == 0) {  // fc3 8->1 + sigmoid
    float a = b_fc3[0];
#pragma unroll
    for (int k = 0; k < 8; ++k) a += w_fc3[k] * sV2[k];
    out[64 + p] = 1.f / (1.f + __expf(-a));
  }
}

// ---------------- launch ----------------
extern "C" void kernel_launch(void* const* d_in, const int* in_sizes, int n_in,
                              void* d_out, int out_size, void* d_ws, size_t ws_size,
                              hipStream_t stream) {
  const float* x        = (const float*)d_in[0];
  const float* w_conv1d = (const float*)d_in[1];
  const float* b_conv1d = (const float*)d_in[2];
  const float* w_c1     = (const float*)d_in[3];
  const float* b_c1     = (const float*)d_in[4];
  const float* w_c2     = (const float*)d_in[5];
  const float* b_c2     = (const float*)d_in[6];
  const float* w_fc1    = (const float*)d_in[7];
  const float* b_fc1    = (const float*)d_in[8];
  const float* w_fc2    = (const float*)d_in[9];
  const float* b_fc2    = (const float*)d_in[10];
  const float* w_fc3    = (const float*)d_in[11];
  const float* b_fc3    = (const float*)d_in[12];
  const float* w_mlp    = (const float*)d_in[13];
  const float* b_mlp    = (const float*)d_in[14];
  float* out = (float*)d_out;

  unsigned char* ws = (unsigned char*)d_ws;
  unsigned short* wA = (unsigned short*)(ws);                 // 256*512 bf16   (256 KB)
  unsigned short* wC = (unsigned short*)(ws + 262144);        // 64*256 bf16    (32 KB)
  unsigned short* xT = (unsigned short*)(ws + 294912);        // 64*64*512 bf16 (4 MB)
  unsigned short* F  = (unsigned short*)(ws + 4489216);       // 64*256*64 bf16 (2 MB)
  unsigned short* G  = (unsigned short*)(ws + 6586368);       // 64*64*64 bf16  (512 KB)

  k0_prep<<<256, 256, 0, stream>>>(x, w_conv1d, w_c1, wA, wC, xT);
  k1_feat<<<NS, 256, 65536 + 32768, stream>>>(xT, b_conv1d, wA, wC, F, G);
  k2_score<<<NS, 256, 0, stream>>>(x, w_mlp, b_mlp, out);
  k3_pair<<<PP, 256, 16384 + 8192 + 32768, stream>>>(F, G, b_c1, w_c2, b_c2,
                                                     w_fc1, b_fc1, w_fc2, b_fc2,
                                                     w_fc3, b_fc3, out);
}
